// LagunaMoE_27462020891146
// MI455X (gfx1250) — matrix operand order repaired and checked by reading, not compiled, not run on hardware
//
#include <hip/hip_runtime.h>

// Problem constants (match reference)
#define T_TOK 2048
#define H_DIM 2048
#define E_EXP 16
#define K_TOP 4
#define I_DIM 1024
#define S_DIM 4096
#define SOFTCAP 30.0f

typedef __bf16 bf16_t;
typedef bf16_t v16bf __attribute__((ext_vector_type(16)));
typedef bf16_t v2bf  __attribute__((ext_vector_type(2)));
typedef float  v8f   __attribute__((ext_vector_type(8)));
typedef float  v2f   __attribute__((ext_vector_type(2)));

#define BM 128
#define BN 128
#define BK 32
#define LPITCH 48   // halves per LDS row: 32 data + 16 pad, 96B keeps uint4 alignment

union Frag16 { uint4 u[2]; v16bf v; };

// Scalar bf16 convert (RNE) for epilogues.
__device__ __forceinline__ unsigned short f2bf(float f) {
  union { bf16_t h; unsigned short u; } cv;
  cv.h = (bf16_t)f;
  return cv.u;
}

// Packed pair convert -> single v_cvt_pk_bf16_f32 (both halves in one VGPR).
__device__ __forceinline__ unsigned cvt2(float x, float y) {
  v2f f; f[0] = x; f[1] = y;
  v2bf h = __builtin_convertvector(f, v2bf);
  union { v2bf h; unsigned u; } cv;
  cv.h = h;
  return cv.u;
}

__device__ __forceinline__ v8f zero8() {
  v8f z;
#pragma unroll
  for (int i = 0; i < 8; ++i) z[i] = 0.0f;
  return z;
}

// A fragment, 16x32 bf16 tile (ISA 7.12.2): lanes 0-15 -> K 0..7 & 16..23,
// lanes 16-31 -> K 8..15 & 24..31, row = lane&15.
__device__ __forceinline__ v16bf frag_a(const unsigned short* rowp, int laneHalf) {
  const uint4* p = (const uint4*)rowp;
  Frag16 f; f.u[0] = p[laneHalf]; f.u[1] = p[2 + laneHalf];
  return f.v;
}
// B fragment, 32x16 bf16: lanes 0-15 -> K 0..7 & 16..23, lanes 16-31 -> K 8..15 & 24..31, col = lane&15.
__device__ __forceinline__ v16bf frag_b(const unsigned short* colp, int laneHalf) {
  const uint4* p = (const uint4*)colp;
  Frag16 f; f.u[0] = p[laneHalf]; f.u[1] = p[2 + laneHalf];
  return f.v;
}

// Async global->LDS 16B copy per lane (CDNA5, ASYNCcnt-tracked, no VGPR data).
__device__ __forceinline__ void async_b128(const unsigned short* lds,
                                           const unsigned short* g) {
  unsigned l = (unsigned)(reinterpret_cast<size_t>(lds));   // low 32b = LDS offset
  unsigned long long a = (unsigned long long)(reinterpret_cast<size_t>(g));
  asm volatile("global_load_async_to_lds_b128 %0, %1, off"
               :: "v"(l), "v"(a) : "memory");
}
__device__ __forceinline__ void wait_async_le1() {
  asm volatile("s_wait_asynccnt 0x1" ::: "memory");
}
__device__ __forceinline__ void wait_async_le0() {
  asm volatile("s_wait_asynccnt 0x0" ::: "memory");
}

// 8 fp32 -> 8 bf16 via 4 packed converts, one ds_store_b128.
__device__ __forceinline__ void pack_store(unsigned short* dst, float4 a, float4 b) {
  uint4 pk;
  pk.x = cvt2(a.x, a.y);
  pk.y = cvt2(a.z, a.w);
  pk.z = cvt2(b.x, b.y);
  pk.w = cvt2(b.z, b.w);
  *(uint4*)dst = pk;
}

// ---------------------------------------------------------------- kernels ---

__global__ void cvt_bf16_kernel(const float* __restrict__ x,
                                unsigned short* __restrict__ xb, int n) {
  // n is a multiple of 2; convert pairs with packed cvt.
  int i = blockIdx.x * blockDim.x + threadIdx.x;
  int half = n >> 1;
  unsigned* out = (unsigned*)xb;
  for (; i < half; i += gridDim.x * blockDim.x) {
    float2 f = *((const float2*)x + i);
    out[i] = cvt2(f.x, f.y);
  }
}

// One block (128 threads) per token: logits -> softcap -> sigmoid -> top4 ->
// renorm; append token to each selected expert's list.
__global__ __launch_bounds__(128)
void router_kernel(const float* __restrict__ x, const float* __restrict__ gw,
                   int* __restrict__ ex_cnt, int* __restrict__ ex_tok,
                   float* __restrict__ ex_w, int* __restrict__ ex_slot) {
  __shared__ float part[E_EXP][8];
  __shared__ float sc[E_EXP];
  const int t = blockIdx.x;
  const int tid = threadIdx.x;
  const int e = tid >> 3, l = tid & 7;
  float s = 0.0f;
  const float* xr = x + (size_t)t * H_DIM;
  const float* gr = gw + (size_t)e * H_DIM;
  for (int j = l; j < H_DIM; j += 8) s += xr[j] * gr[j];
  part[e][l] = s;
  __syncthreads();
  if (tid < E_EXP) {
    float v = 0.0f;
#pragma unroll
    for (int j = 0; j < 8; ++j) v += part[tid][j];
    v = tanhf(v / SOFTCAP) * SOFTCAP;
    sc[tid] = 1.0f / (1.0f + expf(-v));
  }
  __syncthreads();
  if (tid == 0) {
    float ls[E_EXP];
#pragma unroll
    for (int i = 0; i < E_EXP; ++i) ls[i] = sc[i];
    int sel[K_TOP]; float val[K_TOP]; float sum = 0.0f;
#pragma unroll
    for (int k = 0; k < K_TOP; ++k) {
      int bi = 0; float bv = ls[0];
      for (int i = 1; i < E_EXP; ++i) if (ls[i] > bv) { bv = ls[i]; bi = i; }
      sel[k] = bi; val[k] = bv; sum += bv; ls[bi] = -1.0f;
    }
    float inv = 1.0f / sum;
#pragma unroll
    for (int k = 0; k < K_TOP; ++k) {
      int ei = sel[k];
      int pos = atomicAdd(&ex_cnt[ei], 1);
      ex_tok[ei * T_TOK + pos]  = t;
      ex_w[ei * T_TOK + pos]    = val[k] * inv;
      ex_slot[ei * T_TOK + pos] = k;
    }
  }
}

// Fused up-projection: Hout[row] = silu(A @ W1^T) * (A @ W3^T), bf16 out.
// gather!=0: per-expert token gather, out row = token*K_TOP+slot.
__global__ __launch_bounds__(512)
void moe_up_kernel(const unsigned short* __restrict__ Xbf,    // [T,Kd] bf16
                   const float* __restrict__ W1,              // [e][N][Kd] fp32
                   const float* __restrict__ W3,
                   unsigned short* __restrict__ Hout,         // [rows][N] bf16
                   const int* __restrict__ ex_tok, const int* __restrict__ ex_slot,
                   const int* __restrict__ ex_cnt,
                   int N, int Kd, int gather) {
  __shared__ unsigned short As[2][BM * LPITCH];   // double-buffered (async filled)
  __shared__ unsigned short B1s[BN * LPITCH];
  __shared__ unsigned short B3s[BN * LPITCH];

  const int e     = blockIdx.z;
  const int mBase = blockIdx.y * BM;
  const int nBase = blockIdx.x * BN;
  const int cnt   = gather ? ex_cnt[e] : T_TOK;
  if (mBase >= cnt) return;

  const float* w1 = W1 + (size_t)e * N * Kd + (size_t)nBase * Kd;
  const float* w3 = W3 + (size_t)e * N * Kd + (size_t)nBase * Kd;

  const int tid = threadIdx.x;
  const int lane = tid & 31, wv = tid >> 5;
  const int wy = wv >> 2, wx = wv & 3;
  const int laneHalf = lane >> 4, lane15 = lane & 15;
  const int frow = tid >> 2, fchk = tid & 3;     // LDS staging assignment

  int aSrcRow;
  {
    int r = mBase + frow;
    if (gather) {
      int rc = (r < cnt) ? r : (cnt - 1);
      aSrcRow = ex_tok[e * T_TOK + rc];
    } else {
      aSrcRow = r;
    }
  }

  const unsigned short* aSrc = Xbf + (size_t)aSrcRow * Kd + fchk * 8;
  const float* w1p = w1 + (size_t)frow * Kd + fchk * 8;
  const float* w3p = w3 + (size_t)frow * Kd + fchk * 8;
  unsigned short* aDst[2] = { &As[0][frow * LPITCH + fchk * 8],
                              &As[1][frow * LPITCH + fchk * 8] };
  unsigned short* b1Dst = B1s + frow * LPITCH + fchk * 8;
  unsigned short* b3Dst = B3s + frow * LPITCH + fchk * 8;

  v8f acc1[2][2], acc3[2][2];
#pragma unroll
  for (int i = 0; i < 2; ++i)
#pragma unroll
    for (int j = 0; j < 2; ++j) { acc1[i][j] = zero8(); acc3[i][j] = zero8(); }

  const int ksteps = Kd / BK;

  // Prologue: issue A(0) async copy, load B(0) into registers.
  async_b128(aDst[0], aSrc);
  float4 c10 = *(const float4*)(w1p);
  float4 c11 = *(const float4*)(w1p + 4);
  float4 c30 = *(const float4*)(w3p);
  float4 c31 = *(const float4*)(w3p + 4);

#pragma unroll 2
  for (int kt = 0; kt < ksteps; ++kt) {
    // Store current B registers (tile kt) to LDS with packed fp32->bf16.
    pack_store(b1Dst, c10, c11);
    pack_store(b3Dst, c30, c31);

    const bool more = (kt + 1 < ksteps);
    float4 n10, n11, n30, n31;
    if (more) {
      const float* s1 = w1p + (kt + 1) * BK;
      const float* s3 = w3p + (kt + 1) * BK;
      n10 = *(const float4*)(s1);
      n11 = *(const float4*)(s1 + 4);
      n30 = *(const float4*)(s3);
      n31 = *(const float4*)(s3 + 4);
      async_b128(aDst[(kt + 1) & 1], aSrc + (size_t)(kt + 1) * BK);
      wait_async_le1();                 // A(kt) landed; A(kt+1) in flight
    } else {
      wait_async_le0();
    }
    __syncthreads();

    const unsigned short* Acur = As[kt & 1];
    v16bf a[2], b1[2], b3[2];
#pragma unroll
    for (int mt = 0; mt < 2; ++mt)
      a[mt] = frag_a(Acur + (wy * 32 + mt * 16 + lane15) * LPITCH, laneHalf);
#pragma unroll
    for (int nt = 0; nt < 2; ++nt) {
      b1[nt] = frag_b(B1s + (wx * 32 + nt * 16 + lane15) * LPITCH, laneHalf);
      b3[nt] = frag_b(B3s + (wx * 32 + nt * 16 + lane15) * LPITCH, laneHalf);
    }
#pragma unroll
    for (int mt = 0; mt < 2; ++mt)
#pragma unroll
      for (int nt = 0; nt < 2; ++nt) {
        acc1[mt][nt] = __builtin_amdgcn_wmma_f32_16x16x32_bf16(
            false, a[mt], false, b1[nt], (short)0, acc1[mt][nt], false, false);
        acc3[mt][nt] = __builtin_amdgcn_wmma_f32_16x16x32_bf16(
            false, a[mt], false, b3[nt], (short)0, acc3[mt][nt], false, false);
      }
    __syncthreads();
    if (more) { c10 = n10; c11 = n11; c30 = n30; c31 = n31; }
  }

  // Epilogue: h = silu(g) * u, scatter rows to (token*K_TOP + slot)
#pragma unroll
  for (int mt = 0; mt < 2; ++mt)
#pragma unroll
    for (int nt = 0; nt < 2; ++nt)
#pragma unroll
      for (int r = 0; r < 8; ++r) {
        int m  = wy * 32 + mt * 16 + laneHalf * 8 + r;
        int gm = mBase + m;
        if (gather && gm >= cnt) continue;
        int orow;
        if (gather) {
          int s = e * T_TOK + gm;
          orow = ex_tok[s] * K_TOP + ex_slot[s];
        } else {
          orow = gm;
        }
        int n  = nBase + wx * 32 + nt * 16 + lane15;
        float g = acc1[mt][nt][r];
        float u = acc3[mt][nt][r];
        float h = g / (1.0f + __expf(-g)) * u;
        Hout[(size_t)orow * N + n] = f2bf(h);
      }
}

// Expert down-projection: RSlot[token*K_TOP+slot][:] = w * (h @ W2[e]^T)
__global__ __launch_bounds__(512)
void moe_down_kernel(const unsigned short* __restrict__ Hbf,  // [T*K][I] bf16
                     const float* __restrict__ W2,            // [E][H][I] fp32
                     float* __restrict__ RSlot,               // [T*K][H] fp32
                     const int* __restrict__ ex_tok, const int* __restrict__ ex_slot,
                     const int* __restrict__ ex_cnt, const float* __restrict__ ex_w) {
  __shared__ unsigned short As[2][BM * LPITCH];
  __shared__ unsigned short Bs[BN * LPITCH];

  const int e     = blockIdx.z;
  const int mBase = blockIdx.y * BM;
  const int nBase = blockIdx.x * BN;
  const int cnt   = ex_cnt[e];
  if (mBase >= cnt) return;

  const float* w2 = W2 + (size_t)e * H_DIM * I_DIM + (size_t)nBase * I_DIM;

  const int tid = threadIdx.x;
  const int lane = tid & 31, wv = tid >> 5;
  const int wy = wv >> 2, wx = wv & 3;
  const int laneHalf = lane >> 4, lane15 = lane & 15;
  const int frow = tid >> 2, fchk = tid & 3;

  int aSrcRow;
  {
    int r = mBase + frow;
    int rc = (r < cnt) ? r : (cnt - 1);
    int s = e * T_TOK + rc;
    aSrcRow = ex_tok[s] * K_TOP + ex_slot[s];
  }

  const unsigned short* aSrc = Hbf + (size_t)aSrcRow * I_DIM + fchk * 8;
  const float* w2p = w2 + (size_t)frow * I_DIM + fchk * 8;
  unsigned short* aDst[2] = { &As[0][frow * LPITCH + fchk * 8],
                              &As[1][frow * LPITCH + fchk * 8] };
  unsigned short* bDst = Bs + frow * LPITCH + fchk * 8;

  v8f acc[2][2];
#pragma unroll
  for (int i = 0; i < 2; ++i)
#pragma unroll
    for (int j = 0; j < 2; ++j) acc[i][j] = zero8();

  const int ksteps = I_DIM / BK;

  async_b128(aDst[0], aSrc);
  float4 c0 = *(const float4*)(w2p);
  float4 c1 = *(const float4*)(w2p + 4);

#pragma unroll 2
  for (int kt = 0; kt < ksteps; ++kt) {
    pack_store(bDst, c0, c1);
    const bool more = (kt + 1 < ksteps);
    float4 n0, n1;
    if (more) {
      const float* s = w2p + (kt + 1) * BK;
      n0 = *(const float4*)(s);
      n1 = *(const float4*)(s + 4);
      async_b128(aDst[(kt + 1) & 1], aSrc + (size_t)(kt + 1) * BK);
      wait_async_le1();
    } else {
      wait_async_le0();
    }
    __syncthreads();

    const unsigned short* Acur = As[kt & 1];
    v16bf a[2], b[2];
#pragma unroll
    for (int mt = 0; mt < 2; ++mt)
      a[mt] = frag_a(Acur + (wy * 32 + mt * 16 + lane15) * LPITCH, laneHalf);
#pragma unroll
    for (int nt = 0; nt < 2; ++nt)
      b[nt] = frag_b(Bs + (wx * 32 + nt * 16 + lane15) * LPITCH, laneHalf);
#pragma unroll
    for (int mt = 0; mt < 2; ++mt)
#pragma unroll
      for (int nt = 0; nt < 2; ++nt)
        acc[mt][nt] = __builtin_amdgcn_wmma_f32_16x16x32_bf16(
            false, a[mt], false, b[nt], (short)0, acc[mt][nt], false, false);
    __syncthreads();
    if (more) { c0 = n0; c1 = n1; }
  }

#pragma unroll
  for (int mt = 0; mt < 2; ++mt)
#pragma unroll
    for (int nt = 0; nt < 2; ++nt)
#pragma unroll
      for (int r = 0; r < 8; ++r) {
        int m  = wy * 32 + mt * 16 + laneHalf * 8 + r;
        int gm = mBase + m;
        if (gm >= cnt) continue;
        int s    = e * T_TOK + gm;
        int orow = ex_tok[s] * K_TOP + ex_slot[s];
        float w  = ex_w[s];
        int n    = nBase + wx * 32 + nt * 16 + lane15;
        RSlot[(size_t)orow * H_DIM + n] = w * acc[mt][nt][r];
      }
}

// Shared down-projection + final combine: out = hs @ sw2^T + sum_k RSlot[t*K+k]
__global__ __launch_bounds__(512)
void moe_final_kernel(const unsigned short* __restrict__ Hsbf, // [T][S] bf16
                      const float* __restrict__ SW2,           // [H][S] fp32
                      const float* __restrict__ RSlot,         // [T*K][H] fp32
                      float* __restrict__ Out) {               // [T][H] fp32
  __shared__ unsigned short As[2][BM * LPITCH];
  __shared__ unsigned short Bs[BN * LPITCH];

  const int mBase = blockIdx.y * BM;
  const int nBase = blockIdx.x * BN;
  const float* w = SW2 + (size_t)nBase * S_DIM;

  const int tid = threadIdx.x;
  const int lane = tid & 31, wv = tid >> 5;
  const int wy = wv >> 2, wx = wv & 3;
  const int laneHalf = lane >> 4, lane15 = lane & 15;
  const int frow = tid >> 2, fchk = tid & 3;

  const unsigned short* aSrc = Hsbf + (size_t)(mBase + frow) * S_DIM + fchk * 8;
  const float* wp = w + (size_t)frow * S_DIM + fchk * 8;
  unsigned short* aDst[2] = { &As[0][frow * LPITCH + fchk * 8],
                              &As[1][frow * LPITCH + fchk * 8] };
  unsigned short* bDst = Bs + frow * LPITCH + fchk * 8;

  v8f acc[2][2];
#pragma unroll
  for (int i = 0; i < 2; ++i)
#pragma unroll
    for (int j = 0; j < 2; ++j) acc[i][j] = zero8();

  const int ksteps = S_DIM / BK;

  async_b128(aDst[0], aSrc);
  float4 c0 = *(const float4*)(wp);
  float4 c1 = *(const float4*)(wp + 4);

#pragma unroll 2
  for (int kt = 0; kt < ksteps; ++kt) {
    pack_store(bDst, c0, c1);
    const bool more = (kt + 1 < ksteps);
    float4 n0, n1;
    if (more) {
      const float* s = wp + (kt + 1) * BK;
      n0 = *(const float4*)(s);
      n1 = *(const float4*)(s + 4);
      async_b128(aDst[(kt + 1) & 1], aSrc + (size_t)(kt + 1) * BK);
      wait_async_le1();
    } else {
      wait_async_le0();
    }
    __syncthreads();

    const unsigned short* Acur = As[kt & 1];
    v16bf a[2], b[2];
#pragma unroll
    for (int mt = 0; mt < 2; ++mt)
      a[mt] = frag_a(Acur + (wy * 32 + mt * 16 + lane15) * LPITCH, laneHalf);
#pragma unroll
    for (int nt = 0; nt < 2; ++nt)
      b[nt] = frag_b(Bs + (wx * 32 + nt * 16 + lane15) * LPITCH, laneHalf);
#pragma unroll
    for (int mt = 0; mt < 2; ++mt)
#pragma unroll
      for (int nt = 0; nt < 2; ++nt)
        acc[mt][nt] = __builtin_amdgcn_wmma_f32_16x16x32_bf16(
            false, a[mt], false, b[nt], (short)0, acc[mt][nt], false, false);
    __syncthreads();
    if (more) { c0 = n0; c1 = n1; }
  }

#pragma unroll
  for (int mt = 0; mt < 2; ++mt)
#pragma unroll
    for (int nt = 0; nt < 2; ++nt)
#pragma unroll
      for (int r = 0; r < 8; ++r) {
        int t = mBase + wy * 32 + mt * 16 + laneHalf * 8 + r;
        int n = nBase + wx * 32 + nt * 16 + lane15;
        float v = acc[mt][nt][r];
        const float* rs = RSlot + ((size_t)t * K_TOP) * H_DIM + n;
        v += rs[0] + rs[(size_t)H_DIM] + rs[2 * (size_t)H_DIM] + rs[3 * (size_t)H_DIM];
        Out[(size_t)t * H_DIM + n] = v;
      }
}

// ---------------------------------------------------------------- launch ----

extern "C" void kernel_launch(void* const* d_in, const int* in_sizes, int n_in,
                              void* d_out, int out_size, void* d_ws, size_t ws_size,
                              hipStream_t stream) {
  const float* x      = (const float*)d_in[0];
  const float* gate_w = (const float*)d_in[1];
  const float* w1     = (const float*)d_in[2];
  const float* w3     = (const float*)d_in[3];
  const float* w2     = (const float*)d_in[4];
  const float* sw1    = (const float*)d_in[5];
  const float* sw3    = (const float*)d_in[6];
  const float* sw2    = (const float*)d_in[7];
  float* out = (float*)d_out;

  // Workspace carve-up (~110 MB)
  char* ws = (char*)d_ws;
  int* ex_cnt = (int*)ws;                 ws += 256;
  int* ex_tok = (int*)ws;                 ws += (size_t)E_EXP * T_TOK * 4;
  int* ex_slot = (int*)ws;                ws += (size_t)E_EXP * T_TOK * 4;
  float* ex_w = (float*)ws;               ws += (size_t)E_EXP * T_TOK * 4;
  unsigned short* x_bf = (unsigned short*)ws;  ws += (size_t)T_TOK * H_DIM * 2;
  unsigned short* h_ws = (unsigned short*)ws;  ws += (size_t)T_TOK * K_TOP * I_DIM * 2;
  unsigned short* hs_ws = (unsigned short*)ws; ws += (size_t)T_TOK * S_DIM * 2;
  float* rslot = (float*)ws;              ws += (size_t)T_TOK * K_TOP * H_DIM * 4;

  hipMemsetAsync(ex_cnt, 0, E_EXP * sizeof(int), stream);

  cvt_bf16_kernel<<<512, 256, 0, stream>>>(x, x_bf, T_TOK * H_DIM);

  router_kernel<<<T_TOK, 128, 0, stream>>>(x, gate_w, ex_cnt, ex_tok, ex_w, ex_slot);

  // Routed experts: up (gathered) -> h
  moe_up_kernel<<<dim3(I_DIM / BN, T_TOK / BM, E_EXP), 512, 0, stream>>>(
      x_bf, w1, w3, h_ws, ex_tok, ex_slot, ex_cnt, I_DIM, H_DIM, 1);

  // Shared expert: up (dense) -> hs
  moe_up_kernel<<<dim3(S_DIM / BN, T_TOK / BM, 1), 512, 0, stream>>>(
      x_bf, sw1, sw3, hs_ws, nullptr, nullptr, nullptr, S_DIM, H_DIM, 0);

  // Routed experts: down, weighted, scattered to unique slot rows
  moe_down_kernel<<<dim3(H_DIM / BN, T_TOK / BM, E_EXP), 512, 0, stream>>>(
      h_ws, w2, rslot, ex_tok, ex_slot, ex_cnt, ex_w);

  // Shared down + combine 4 routed slots -> out
  moe_final_kernel<<<dim3(H_DIM / BN, T_TOK / BM, 1), 512, 0, stream>>>(
      hs_ws, sw2, rslot, out);
}
